// Attention_54752243089781
// MI455X (gfx1250) — matrix operand order repaired and checked by reading, not compiled, not run on hardware
//
#include <hip/hip_runtime.h>
#include <hip/hip_bf16.h>

typedef __attribute__((ext_vector_type(16))) __bf16 v16bf;
typedef __attribute__((ext_vector_type(2)))  __bf16 v2bf;
typedef __attribute__((ext_vector_type(8)))  float  v8f;

union FragBF {
    uint4 q[2];
    unsigned u[8];
    v16bf v;
};

__device__ __forceinline__ unsigned short f32_to_bf16_rne(float f) {
    unsigned u = __builtin_bit_cast(unsigned, f);
    unsigned r = 0x7FFFu + ((u >> 16) & 1u);
    return (unsigned short)((u + r) >> 16);
}

__device__ __forceinline__ unsigned pack_bf16x2(float lo, float hi) {
#if __has_builtin(__builtin_amdgcn_cvt_pk_bf16_f32)
    v2bf p = __builtin_amdgcn_cvt_pk_bf16_f32(lo, hi);
    return __builtin_bit_cast(unsigned, p);
#else
    return (unsigned)f32_to_bf16_rne(lo) | ((unsigned)f32_to_bf16_rne(hi) << 16);
#endif
}

// ---------------------------------------------------------------------------
// f32 -> bf16 bulk conversion, 4 elements / thread / step
// ---------------------------------------------------------------------------
__global__ __launch_bounds__(256) void cvt_f32_bf16x4_kernel(
    const float4* __restrict__ src, uint2* __restrict__ dst, int n4) {
    int i = blockIdx.x * blockDim.x + threadIdx.x;
    int stride = gridDim.x * blockDim.x;
    for (; i < n4; i += stride) {
        float4 v = src[i];
        uint2 o;
        o.x = pack_bf16x2(v.x, v.y);
        o.y = pack_bf16x2(v.z, v.w);
        dst[i] = o;
    }
}

// ---------------------------------------------------------------------------
// bf16 WMMA GEMM:  C(MxN) = A(MxK) * B(KxN), epilogue (acc + bias[col]) * scale
// BM=128 BN=128 BK=32, 256 threads = 8 waves in 4x2 grid, each wave 32x64.
// Double-buffered LDS staging: global loads for tile k+1 issued before the
// 8 WMMAs of tile k; one barrier per K-step; prefetch 2 tiles ahead.
// ---------------------------------------------------------------------------
#define BM 128
#define BN 128
#define BK 32
#define LDA 40   // padded LDS row stride in bf16 elements (80B, 16B aligned)

template <bool OUT_BF16>
__global__ __launch_bounds__(256) void gemm_bf16_kernel(
    const unsigned short* __restrict__ A,   // M x K, row-major bf16
    const unsigned short* __restrict__ B,   // K x N, row-major bf16
    const float* __restrict__ bias,         // N
    void* __restrict__ Cout,                // M x N (bf16 or f32)
    int M, int N, int K, float scale) {
    __shared__ unsigned short As[2][BM * LDA];
    __shared__ unsigned short Bs[2][BN * LDA]; // transposed: [n][k]

    const int tid  = threadIdx.x;
    const int lane = tid & 31;
    const int wid  = tid >> 5;
    const int wm   = wid & 3;    // 32-row strip
    const int wn   = wid >> 2;   // 64-col strip
    const int m0   = blockIdx.y * BM;
    const int n0   = blockIdx.x * BN;

    v8f acc[2][4];
#pragma unroll
    for (int mt = 0; mt < 2; ++mt)
#pragma unroll
        for (int nt = 0; nt < 4; ++nt) acc[mt][nt] = {};

    const int l16 = lane & 15;
    const int akb = (lane < 16) ? 0 : 8;    // A K-offset (elements)
    const int bkb = (lane < 16) ? 0 : 8;    // B K-offset (elements)

    // per-thread staging coordinates
    const int sar = tid >> 1;               // A row 0..127
    const int sac = (tid & 1) * 16;         // A col 0/16
    const int sbk = tid >> 3;               // B k-row 0..31
    const int sbn = (tid & 7) * 16;         // B n-col base
    const unsigned short* Ag = A + (size_t)(m0 + sar) * K + sac;
    const unsigned short* Bg = B + (size_t)sbk * N + n0 + sbn;

    // prologue: stage tile 0 into buffer 0
    uint4 pa0 = *(const uint4*)(Ag);
    uint4 pa1 = *(const uint4*)(Ag + 8);
    uint4 pb0 = *(const uint4*)(Bg);
    uint4 pb1 = *(const uint4*)(Bg + 8);
    *(uint4*)&As[0][sar * LDA + sac]     = pa0;
    *(uint4*)&As[0][sar * LDA + sac + 8] = pa1;
    {
        const unsigned short* s0 = (const unsigned short*)&pb0;
        const unsigned short* s1 = (const unsigned short*)&pb1;
#pragma unroll
        for (int e = 0; e < 8; ++e) Bs[0][(sbn + e) * LDA + sbk] = s0[e];
#pragma unroll
        for (int e = 0; e < 8; ++e) Bs[0][(sbn + 8 + e) * LDA + sbk] = s1[e];
    }
    __syncthreads();

    const int NK = K / BK;
    for (int kt = 0; kt < NK; ++kt) {
        const int cur = kt & 1;
        const bool has_next = (kt + 1) < NK;
        if (has_next) {
            const unsigned short* ga = Ag + (kt + 1) * BK;
            const unsigned short* gb = Bg + (size_t)(kt + 1) * BK * N;
            pa0 = *(const uint4*)(ga);
            pa1 = *(const uint4*)(ga + 8);
            pb0 = *(const uint4*)(gb);
            pb1 = *(const uint4*)(gb + 8);
            if (kt + 2 < NK) {
                __builtin_prefetch(Ag + (kt + 2) * BK, 0, 0);
                __builtin_prefetch(Bg + (size_t)(kt + 2) * BK * N, 0, 0);
            }
        }

        // compute from current buffer
        const unsigned short* as = As[cur];
        const unsigned short* bs = Bs[cur];
        FragBF a[2], bfr[4];
#pragma unroll
        for (int mt = 0; mt < 2; ++mt) {
            const unsigned short* p = &as[(wm * 32 + mt * 16 + l16) * LDA];
            a[mt].q[0] = *(const uint4*)(p + akb);
            a[mt].q[1] = *(const uint4*)(p + 16 + akb);
        }
#pragma unroll
        for (int nt = 0; nt < 4; ++nt) {
            const unsigned short* p = &bs[(wn * 64 + nt * 16 + l16) * LDA + bkb];
            bfr[nt].q[0] = *(const uint4*)p;
            bfr[nt].q[1] = *(const uint4*)(p + 16);
        }
#pragma unroll
        for (int mt = 0; mt < 2; ++mt)
#pragma unroll
            for (int nt = 0; nt < 4; ++nt)
                acc[mt][nt] = __builtin_amdgcn_wmma_f32_16x16x32_bf16(
                    false, a[mt].v, false, bfr[nt].v, (short)0, acc[mt][nt],
                    false, false);

        // stage tile kt+1 into the other buffer
        if (has_next) {
            const int nxt = cur ^ 1;
            *(uint4*)&As[nxt][sar * LDA + sac]     = pa0;
            *(uint4*)&As[nxt][sar * LDA + sac + 8] = pa1;
            const unsigned short* s0 = (const unsigned short*)&pb0;
            const unsigned short* s1 = (const unsigned short*)&pb1;
#pragma unroll
            for (int e = 0; e < 8; ++e) Bs[nxt][(sbn + e) * LDA + sbk] = s0[e];
#pragma unroll
            for (int e = 0; e < 8; ++e) Bs[nxt][(sbn + 8 + e) * LDA + sbk] = s1[e];
        }
        __syncthreads();
    }

    const int rbase = (lane >> 4) * 8;
#pragma unroll
    for (int nt = 0; nt < 4; ++nt) {
        int col = n0 + wn * 64 + nt * 16 + l16;
        float bv = bias[col];
#pragma unroll
        for (int mt = 0; mt < 2; ++mt) {
#pragma unroll
            for (int r = 0; r < 8; ++r) {
                int row = m0 + wm * 32 + mt * 16 + rbase + r;
                float val = (acc[mt][nt][r] + bv) * scale;
                if (OUT_BF16)
                    ((unsigned short*)Cout)[(size_t)row * N + col] =
                        f32_to_bf16_rne(val);
                else
                    ((float*)Cout)[(size_t)row * N + col] = val;
            }
        }
    }
}

// ---------------------------------------------------------------------------
// Fused attention per (b, head): logits=QK^T+bias, softmax, out=P*V.
// Full 256x256 f32 logits tile resident in LDS (CDNA5: 320KB/WGP).
// ---------------------------------------------------------------------------
#define SEQ 256
#define HS  32
#define QLD 40    // q_s / k_s row stride (bf16 elems)
#define VLD 264   // v_t row stride (bf16 elems)
#define LLD 260   // logits row stride (f32 elems, %4==0 for float4)

__global__ __launch_bounds__(256) void attn_kernel(
    const unsigned short* __restrict__ Q,   // (B*S, 512) bf16, col = head*32+d
    const unsigned short* __restrict__ Kc,  // same layout
    const unsigned short* __restrict__ V,   // same layout
    const float* __restrict__ rel_emb,      // (16, 31, 31) f32
    unsigned short* __restrict__ Out) {     // (B*S, 512) bf16
    extern __shared__ char smem[];
    unsigned short* q_s = (unsigned short*)smem;     // SEQ x QLD
    unsigned short* k_s = q_s + SEQ * QLD;           // SEQ x QLD
    unsigned short* v_t = k_s + SEQ * QLD;           // HS x VLD (transposed)
    float* lg = (float*)(v_t + HS * VLD);            // SEQ x LLD

    const int head = blockIdx.x;
    const int b    = blockIdx.y;
    const int tid  = threadIdx.x;
    const int lane = tid & 31;
    const int wid  = tid >> 5;
    const int l16  = lane & 15;
    const int akb  = (lane < 16) ? 0 : 8;
    const int bkb  = (lane < 16) ? 0 : 8;

    // ---- load Q/K rows, V transposed ----
    {
        int s = tid;
        const size_t base = ((size_t)(b * SEQ + s)) * 512 + head * HS;
        const uint4* qg = (const uint4*)(Q + base);
        uint4* qd = (uint4*)&q_s[s * QLD];
        qd[0] = qg[0]; qd[1] = qg[1]; qd[2] = qg[2]; qd[3] = qg[3];
        const uint4* kg = (const uint4*)(Kc + base);
        uint4* kd = (uint4*)&k_s[s * QLD];
        kd[0] = kg[0]; kd[1] = kg[1]; kd[2] = kg[2]; kd[3] = kg[3];
        uint4 vv[4];
        const uint4* vg = (const uint4*)(V + base);
        vv[0] = vg[0]; vv[1] = vg[1]; vv[2] = vg[2]; vv[3] = vg[3];
        const unsigned short* ve = (const unsigned short*)vv;
#pragma unroll
        for (int d = 0; d < HS; ++d) v_t[d * VLD + s] = ve[d];
    }
    __syncthreads();

    // ---- logits = Q K^T + relative bias (K=32 -> one wmma per tile) ----
#pragma unroll 1
    for (int i = 0; i < 32; ++i) {
        int tile = wid * 32 + i;
        int tm = tile >> 4, tn = tile & 15;
        FragBF a, bf;
        const unsigned short* ap = &q_s[(tm * 16 + l16) * QLD];
        a.q[0] = *(const uint4*)(ap + akb);
        a.q[1] = *(const uint4*)(ap + 16 + akb);
        const unsigned short* bp = &k_s[(tn * 16 + l16) * QLD + bkb];
        bf.q[0] = *(const uint4*)bp;
        bf.q[1] = *(const uint4*)(bp + 16);
        v8f c = {};
        c = __builtin_amdgcn_wmma_f32_16x16x32_bf16(false, a.v, false, bf.v,
                                                    (short)0, c, false, false);
        int scol = tn * 16 + l16;        // key index t = (jy, jx)
        int jy = scol >> 4, jx = scol & 15;
#pragma unroll
        for (int r = 0; r < 8; ++r) {
            int srow = tm * 16 + (lane >> 4) * 8 + r;  // query s = (iy, ix)
            int iy = srow >> 4, ix = srow & 15;
            float bias =
                rel_emb[(head * 31 + (jy - iy + 15)) * 31 + (jx - ix + 15)];
            lg[srow * LLD + scol] = c[r] + bias;
        }
    }
    __syncthreads();

    // ---- per-row softmax (one thread per row, float4 vectorized) ----
    {
        float* row = &lg[tid * LLD];
        float m = -3.4e38f;
        for (int j = 0; j < SEQ; j += 4) {
            float4 v = *(const float4*)(row + j);
            m = fmaxf(m, fmaxf(fmaxf(v.x, v.y), fmaxf(v.z, v.w)));
        }
        float sum = 0.f;
        for (int j = 0; j < SEQ; j += 4) {
            float4 v = *(const float4*)(row + j);
            v.x = __expf(v.x - m);
            v.y = __expf(v.y - m);
            v.z = __expf(v.z - m);
            v.w = __expf(v.w - m);
            sum += (v.x + v.y) + (v.z + v.w);
            *(float4*)(row + j) = v;
        }
        float inv = 1.f / sum;
        for (int j = 0; j < SEQ; j += 4) {
            float4 v = *(const float4*)(row + j);
            v.x *= inv; v.y *= inv; v.z *= inv; v.w *= inv;
            *(float4*)(row + j) = v;
        }
    }
    __syncthreads();

    // ---- out = P * V  (M=256, N=32, K=256 -> 8 wmma K-steps per tile) ----
#pragma unroll 1
    for (int i = 0; i < 4; ++i) {
        int tile = wid * 4 + i;
        int mt = tile >> 1, dt = tile & 1;
        v8f acc = {};
        const float* arow = &lg[(mt * 16 + l16) * LLD];
        const unsigned short* brow = &v_t[(dt * 16 + l16) * VLD];
#pragma unroll
        for (int ks = 0; ks < 8; ++ks) {
            FragBF a, bf;
            const float* p = arow + ks * 32 + akb;
            float4 f0 = *(const float4*)(p);
            float4 f1 = *(const float4*)(p + 4);
            float4 f2 = *(const float4*)(p + 16);
            float4 f3 = *(const float4*)(p + 20);
            a.u[0] = pack_bf16x2(f0.x, f0.y);
            a.u[1] = pack_bf16x2(f0.z, f0.w);
            a.u[2] = pack_bf16x2(f1.x, f1.y);
            a.u[3] = pack_bf16x2(f1.z, f1.w);
            a.u[4] = pack_bf16x2(f2.x, f2.y);
            a.u[5] = pack_bf16x2(f2.z, f2.w);
            a.u[6] = pack_bf16x2(f3.x, f3.y);
            a.u[7] = pack_bf16x2(f3.z, f3.w);
            const unsigned short* bp = brow + bkb + ks * 32;
            bf.q[0] = *(const uint4*)bp;
            bf.q[1] = *(const uint4*)(bp + 16);
            acc = __builtin_amdgcn_wmma_f32_16x16x32_bf16(
                false, a.v, false, bf.v, (short)0, acc, false, false);
        }
#pragma unroll
        for (int r = 0; r < 8; ++r) {
            int srow = mt * 16 + (lane >> 4) * 8 + r;
            int d = dt * 16 + l16;
            Out[((size_t)(b * SEQ + srow)) * 512 + head * HS + d] =
                f32_to_bf16_rne(acc[r]);
        }
    }
}

// ---------------------------------------------------------------------------
extern "C" void kernel_launch(void* const* d_in, const int* in_sizes, int n_in,
                              void* d_out, int out_size, void* d_ws,
                              size_t ws_size, hipStream_t stream) {
    (void)in_sizes; (void)n_in; (void)out_size; (void)ws_size;
    const float* x    = (const float*)d_in[0];
    const float* q_w  = (const float*)d_in[1];
    const float* q_b  = (const float*)d_in[2];
    const float* k_w  = (const float*)d_in[3];
    const float* k_b  = (const float*)d_in[4];
    const float* v_w  = (const float*)d_in[5];
    const float* v_b  = (const float*)d_in[6];
    const float* o_w  = (const float*)d_in[7];
    const float* o_b  = (const float*)d_in[8];
    const float* rel  = (const float*)d_in[9];

    const int M = 64 * 256;   // 16384 tokens
    const int C = 512;

    unsigned short* x_bf  = (unsigned short*)d_ws;
    unsigned short* qw_bf = x_bf + (size_t)M * C;
    unsigned short* kw_bf = qw_bf + (size_t)C * C;
    unsigned short* vw_bf = kw_bf + (size_t)C * C;
    unsigned short* ow_bf = vw_bf + (size_t)C * C;
    unsigned short* q_bf  = ow_bf + (size_t)C * C;
    unsigned short* k_bf  = q_bf + (size_t)M * C;
    unsigned short* v_bf  = k_bf + (size_t)M * C;
    unsigned short* ao_bf = v_bf + (size_t)M * C;

    cvt_f32_bf16x4_kernel<<<2048, 256, 0, stream>>>(
        (const float4*)x, (uint2*)x_bf, M * C / 4);
    cvt_f32_bf16x4_kernel<<<256, 256, 0, stream>>>(
        (const float4*)q_w, (uint2*)qw_bf, C * C / 4);
    cvt_f32_bf16x4_kernel<<<256, 256, 0, stream>>>(
        (const float4*)k_w, (uint2*)kw_bf, C * C / 4);
    cvt_f32_bf16x4_kernel<<<256, 256, 0, stream>>>(
        (const float4*)v_w, (uint2*)vw_bf, C * C / 4);
    cvt_f32_bf16x4_kernel<<<256, 256, 0, stream>>>(
        (const float4*)o_w, (uint2*)ow_bf, C * C / 4);

    dim3 g(C / BN, M / BM);
    const float scale = 0.17677669529663687f;  // 32^-0.5
    gemm_bf16_kernel<true><<<g, 256, 0, stream>>>(x_bf, qw_bf, q_b, q_bf, M, C, C, scale);
    gemm_bf16_kernel<true><<<g, 256, 0, stream>>>(x_bf, kw_bf, k_b, k_bf, M, C, C, 1.0f);
    gemm_bf16_kernel<true><<<g, 256, 0, stream>>>(x_bf, vw_bf, v_b, v_bf, M, C, C, 1.0f);

    size_t attn_lds = (size_t)(2 * SEQ * QLD + HS * VLD) * sizeof(unsigned short) +
                      (size_t)SEQ * LLD * sizeof(float);   // 324,096 B <= 320KB+pad
    attn_kernel<<<dim3(16, 64), 256, attn_lds, stream>>>(q_bf, k_bf, v_bf, rel, ao_bf);

    gemm_bf16_kernel<false><<<g, 256, 0, stream>>>(ao_bf, ow_bf, o_b, d_out, M, C, C, 1.0f);
}
